// M6_GraphBiLSTM_81174881894572
// MI455X (gfx1250) — matrix operand order repaired and checked by reading, not compiled, not run on hardware
//
#include <hip/hip_runtime.h>
#include <hip/hip_bf16.h>
#include <hip/hip_fp16.h>

// ---------------------------------------------------------------------------
// Model dims
// ---------------------------------------------------------------------------
#define NB     256          // batch
#define NCH    16           // channels (graph nodes)
#define NTIME  256          // input timesteps
#define NH     128          // hidden H
#define NHEADS 4
#define DHEAD  32           // GAT per-head dim
#define TCONV  128          // timesteps after stride-2 conv
#define NT     (NB*TCONV)   // 32768 (b,t) rows
#define NCROWS (NT*NCH)     // 524288 (b,t,c) rows
#define EMB    256          // 2*H
#define HD     64           // MHA head dim

typedef __attribute__((ext_vector_type(16))) _Float16 v16h;
typedef __attribute__((ext_vector_type(2)))  _Float16 v2h;
typedef __attribute__((ext_vector_type(8)))  float    v8f;

// ---------------------------------------------------------------------------
// WMMA fragment helpers (CDNA5 wave32, v_wmma_f32_16x16x32_f16)
// A 16x32 f16: lane lo = row M, lane hi selects K {0..7,16..23}/{8..15,24..31}
// B 32x16 f16: lane lo = col N, lane hi selects K {0..7,16..23}/{8..15,24..31} (same as A)
// C/D 16x16 f32: VGPR v holds row (v + 8*hi), col = lane lo.
// All B operands are stored N x K row-major (weights pre-transposed), so both
// A and B fragment loads are K-contiguous and coalesce to b128.
// ---------------------------------------------------------------------------
__device__ __forceinline__ v16h wmma_load_a(const _Float16* __restrict__ p /* row base + hi*8 */) {
  v16h a;
#pragma unroll
  for (int q = 0; q < 8; ++q) {
    int kb = ((q & 4) ? 16 : 0) + (q & 3) * 2;
    v2h t = *(const v2h*)(p + kb);
    a[2 * q]     = t[0];
    a[2 * q + 1] = t[1];
  }
  return a;
}
__device__ __forceinline__ v16h wmma_load_b(const _Float16* __restrict__ p /* col base + hi*8 */) {
  v16h b;
#pragma unroll
  for (int q = 0; q < 8; ++q) {
    int kb = ((q & 4) ? 16 : 0) + (q & 3) * 2;
    v2h t = *(const v2h*)(p + kb);
    b[2 * q]     = t[0];
    b[2 * q + 1] = t[1];
  }
  return b;
}
__device__ __forceinline__ v8f wmma_mac(v16h a, v16h b, v8f c) {
  return __builtin_amdgcn_wmma_f32_16x16x32_f16(false, a, false, b, (short)0, c, false, false);
}

// ---------------------------------------------------------------------------
// Direct WMMA GEMM (one wave per 16x16 tile), specialized by template so the
// K loop is pure loads+wmma. D = A * B^T (+Cin) (+bias).
// ---------------------------------------------------------------------------
template <int HASC, int HASBIAS, int OUT32, int OUT16>
__global__ void gemm_wmma_kernel(const _Float16* __restrict__ A, int lda,
                                 const _Float16* __restrict__ Bm, int ldb,
                                 const float* __restrict__ Cin, int ldc,
                                 const float* __restrict__ bias,
                                 float* __restrict__ D, _Float16* __restrict__ D16,
                                 int ldd, int ntn, int K) {
  int tile = blockIdx.x;
  int tm = tile / ntn, tn = tile - tm * ntn;
  int lane = threadIdx.x & 31;
  int lo = lane & 15, hi = lane >> 4;
  const _Float16* pA = A + (size_t)(tm * 16 + lo) * lda + hi * 8;
  const _Float16* pB = Bm + (size_t)(tn * 16 + lo) * ldb + hi * 8;

  v8f c = {};
  if (HASC) {
#pragma unroll
    for (int v = 0; v < 8; ++v)
      c[v] = Cin[(size_t)(tm * 16 + hi * 8 + v) * ldc + tn * 16 + lo];
  }
  for (int k0 = 0; k0 < K; k0 += 32) {
    __builtin_prefetch(pA + 64, 0, 1);   // speculative: global_prefetch_b8
    v16h a = wmma_load_a(pA);
    v16h b = wmma_load_b(pB);
    c = wmma_mac(a, b, c);
    pA += 32;
    pB += 32;
  }
#pragma unroll
  for (int v = 0; v < 8; ++v) {
    int r  = tm * 16 + hi * 8 + v;
    int cc = tn * 16 + lo;
    float val = c[v];
    if (HASBIAS) val += bias[cc];
    if (OUT32) D[(size_t)r * ldd + cc] = val;
    if (OUT16) D16[(size_t)r * ldd + cc] = (_Float16)val;
  }
}

// ---------------------------------------------------------------------------
// LDS-staged double-buffered WMMA GEMM fed by async global->LDS copies.
// Block = 256 threads (8 waves), output tile 32x64, K chunk 64.
// Per chunk each wave issues 3x GLOBAL_LOAD_ASYNC_TO_LDS_B128 (512B each),
// waits with s_wait_asynccnt, barriers, then runs 2 WMMA from LDS.
// Requires lda == ldb == K, M%32==0, N%64==0, K%64==0.
// ---------------------------------------------------------------------------
#define XBM 32
#define XBN 64
#define XKC 64

__global__ void gemm_wmma_async_kernel(const _Float16* __restrict__ A,
                                       const _Float16* __restrict__ Bm,
                                       float* __restrict__ D,
                                       int ldd, int nbn, int K) {
  __shared__ _Float16 sh[2][XBM * XKC + XBN * XKC];   // 2 * 12 KB
  int blk = blockIdx.x;
  int bm = blk / nbn, bn = blk - bm * nbn;
  int tid = threadIdx.x;
  int w = tid >> 5, lane = tid & 31;
  int lo = lane & 15, hi = lane >> 4;
  int wm = w >> 2, wn = w & 3;

  const _Float16* Ab = A + (size_t)(bm * XBM) * K;
  const _Float16* Bb = Bm + (size_t)(bn * XBN) * K;

  // per-thread 16B segment assignment: 256 A segs (1/thread), 512 B segs (2/thread)
  int arow = tid >> 3, aseg = tid & 7;

  auto issue = [&](int p, int k0) {
    unsigned l = (unsigned)(unsigned long long)(&sh[p][arow * XKC + aseg * 8]);
    unsigned long long g = (unsigned long long)(Ab + (size_t)arow * K + k0 + aseg * 8);
    asm volatile("global_load_async_to_lds_b128 %0, %1, off" :: "v"(l), "v"(g) : "memory");
#pragma unroll
    for (int s = 0; s < 2; ++s) {
      int idx = tid + (s << 8);
      int col = idx >> 3, seg = idx & 7;
      unsigned lb = (unsigned)(unsigned long long)(&sh[p][XBM * XKC + col * XKC + seg * 8]);
      unsigned long long gb = (unsigned long long)(Bb + (size_t)col * K + k0 + seg * 8);
      asm volatile("global_load_async_to_lds_b128 %0, %1, off" :: "v"(lb), "v"(gb) : "memory");
    }
  };

  v8f c = {};
  int nK = K / XKC;
  issue(0, 0);
  for (int kc = 0; kc < nK; ++kc) {
    int p = kc & 1;
    if (kc + 1 < nK) {
      issue(p ^ 1, (kc + 1) * XKC);
      asm volatile("s_wait_asynccnt 0x3" ::: "memory");   // current buffer landed
    } else {
      asm volatile("s_wait_asynccnt 0x0" ::: "memory");
    }
    __syncthreads();
    const _Float16* sA = &sh[p][0]           + (size_t)(wm * 16 + lo) * XKC + hi * 8;
    const _Float16* sB = &sh[p][XBM * XKC]   + (size_t)(wn * 16 + lo) * XKC + hi * 8;
#pragma unroll
    for (int kk = 0; kk < 2; ++kk) {
      v16h a = wmma_load_a(sA + kk * 32);
      v16h b = wmma_load_b(sB + kk * 32);
      c = wmma_mac(a, b, c);
    }
    __syncthreads();   // everyone done reading buf p before it is re-filled
  }
  int r0 = bm * XBM + wm * 16 + hi * 8;
  int c0 = bn * XBN + wn * 16 + lo;
#pragma unroll
  for (int v = 0; v < 8; ++v) D[(size_t)(r0 + v) * ldd + c0] = c[v];
}

// ---------------------------------------------------------------------------
// Fused conv1(k7,p3)+BN+GeLU -> conv2(k5,s2,p2)+BN+GeLU, writes f16 GAT layout
// ---------------------------------------------------------------------------
__device__ __forceinline__ float gelu_f(float x) {
  return 0.5f * x * (1.f + erff(x * 0.7071067811865475f));
}
__device__ __forceinline__ float sigm_f(float x) { return 1.f / (1.f + __expf(-x)); }

__global__ void conv_stage_kernel(const float* __restrict__ x,
                                  const float* __restrict__ w1, const float* __restrict__ b1,
                                  const float* __restrict__ g1, const float* __restrict__ be1,
                                  const float* __restrict__ w2, const float* __restrict__ b2,
                                  const float* __restrict__ g2, const float* __restrict__ be2,
                                  _Float16* __restrict__ gout) {
  __shared__ float mid[5][32];
  int blk  = blockIdx.x;
  int tout = blk & (TCONV - 1);
  int bc   = blk >> 7;
  int b    = bc >> 4, c = bc & 15;
  int tid  = threadIdx.x;
  const float* xr = x + (size_t)bc * NTIME;

  for (int idx = tid; idx < 160; idx += 64) {
    int k2 = idx / 32, m = idx - (idx / 32) * 32;
    int t1 = 2 * tout - 2 + k2;
    float acc = 0.f;
    if (t1 >= 0 && t1 < NTIME) {
#pragma unroll
      for (int k1 = 0; k1 < 7; ++k1) {
        int ti = t1 - 3 + k1;
        float xv = (ti >= 0 && ti < NTIME) ? xr[ti] : 0.f;
        acc += xv * w1[m * 7 + k1];
      }
      acc += b1[m];
      acc = gelu_f(acc * g1[m] + be1[m]);
    }
    mid[k2][m] = acc;
  }
  __syncthreads();
  int oc = tid;
  float acc = 0.f;
#pragma unroll
  for (int k2 = 0; k2 < 5; ++k2)
    for (int m = 0; m < 32; ++m)
      acc += mid[k2][m] * w2[(oc * 32 + m) * 5 + k2];
  acc += b2[oc];
  acc = gelu_f(acc * g2[oc] + be2[oc]);
  gout[((size_t)(b * TCONV + tout) * NCH + c) * 64 + oc] = (_Float16)acc;
}

// ---------------------------------------------------------------------------
// GAT attention aggregate (width 128), 64 threads = (head,row), block per (b,t)
// ---------------------------------------------------------------------------
__global__ void gat_attn_kernel(const float* __restrict__ h,
                                const float* __restrict__ asrc, const float* __restrict__ adst,
                                const float* __restrict__ adj, float* __restrict__ out) {
  __shared__ float es[NHEADS][NCH], ed[NHEADS][NCH];
  int n   = blockIdx.x;
  int tid = threadIdx.x;
  int i = tid & 15, hh = tid >> 4;
  const float* hn = h + (size_t)n * NCH * NH;

  const float* hv = hn + i * NH + hh * DHEAD;
  float s1 = 0.f, s2 = 0.f;
#pragma unroll
  for (int d = 0; d < DHEAD; ++d) {
    float v = hv[d];
    s1 += v * asrc[hh * DHEAD + d];
    s2 += v * adst[hh * DHEAD + d];
  }
  es[hh][i] = s1;
  ed[hh][i] = s2;
  __syncthreads();

  float e[NCH];
  float m = -1e30f;
  float esi = es[hh][i];
#pragma unroll
  for (int j = 0; j < NCH; ++j) {
    float t = esi + ed[hh][j];
    t = (t > 0.f) ? t : 0.2f * t;
    t += adj[i * NCH + j];
    e[j] = t;
    m = fmaxf(m, t);
  }
  float ssum = 0.f;
#pragma unroll
  for (int j = 0; j < NCH; ++j) { e[j] = __expf(e[j] - m); ssum += e[j]; }
  float inv = 1.f / ssum;

  float acc[DHEAD];
#pragma unroll
  for (int d = 0; d < DHEAD; ++d) acc[d] = 0.f;
#pragma unroll
  for (int j = 0; j < NCH; ++j) {
    float a = e[j] * inv;
    const float* hj = hn + j * NH + hh * DHEAD;
#pragma unroll
    for (int d = 0; d < DHEAD; ++d) acc[d] += a * hj[d];
  }
  float* op = out + (size_t)n * (NCH * NH) + i * NH + hh * DHEAD;
#pragma unroll
  for (int d = 0; d < DHEAD; ++d) op[d] = acc[d];
}

// ---------------------------------------------------------------------------
// Row LayerNorm (wave32 per row), optional residual/relu, f32/f16 outputs
// ---------------------------------------------------------------------------
__global__ void row_ln_kernel(const float* __restrict__ in, const float* __restrict__ res,
                              const float* __restrict__ gamma, const float* __restrict__ beta,
                              float* __restrict__ out32, _Float16* __restrict__ out16,
                              int W, int do_relu) {
  int row  = blockIdx.x;
  int lane = threadIdx.x & 31;
  int cnt  = W >> 5;
  const float* r = in + (size_t)row * W;
  float buf[16];
  float s = 0.f;
  for (int k = 0; k < cnt; ++k) {
    int idx = lane + (k << 5);
    float v = r[idx];
    if (res) v += res[(size_t)row * W + idx];
    if (do_relu) v = fmaxf(v, 0.f);
    buf[k] = v;
    s += v;
  }
#pragma unroll
  for (int mm = 16; mm >= 1; mm >>= 1) s += __shfl_xor(s, mm, 32);
  float mean = s / (float)W;
  float vs = 0.f;
  for (int k = 0; k < cnt; ++k) { float d = buf[k] - mean; vs += d * d; }
#pragma unroll
  for (int mm = 16; mm >= 1; mm >>= 1) vs += __shfl_xor(vs, mm, 32);
  float rstd = rsqrtf(vs / (float)W + 1e-5f);
  for (int k = 0; k < cnt; ++k) {
    int idx = lane + (k << 5);
    float y = (buf[k] - mean) * rstd * gamma[idx] + beta[idx];
    if (out32) out32[(size_t)row * W + idx] = y;
    if (out16) out16[(size_t)row * W + idx] = (_Float16)y;
  }
}

// ---------------------------------------------------------------------------
// LSTM cell (both directions)
// ---------------------------------------------------------------------------
__global__ void lstm_cell_kernel(const float* __restrict__ gf, const float* __restrict__ gr,
                                 const float* __restrict__ bihf, const float* __restrict__ bhhf,
                                 const float* __restrict__ bihr, const float* __restrict__ bhhr,
                                 float* __restrict__ cf, float* __restrict__ cr,
                                 _Float16* __restrict__ hf16, _Float16* __restrict__ hr16,
                                 float* __restrict__ out32, _Float16* __restrict__ out16, int t) {
  int tid = blockIdx.x * blockDim.x + threadIdx.x;
  int j   = tid & 127;
  int bb  = (tid >> 7) & 255;
  int dir = tid >> 15;
  const float* g   = dir ? gr   : gf;
  const float* bih = dir ? bihr : bihf;
  const float* bhh = dir ? bhhr : bhhf;
  float*    cs = dir ? cr   : cf;
  _Float16* hs = dir ? hr16 : hf16;

  float gi = g[bb * 512 + j]       + bih[j]       + bhh[j];
  float gF = g[bb * 512 + 128 + j] + bih[128 + j] + bhh[128 + j];
  float gg = g[bb * 512 + 256 + j] + bih[256 + j] + bhh[256 + j];
  float go = g[bb * 512 + 384 + j] + bih[384 + j] + bhh[384 + j];
  float cp = cs[bb * 128 + j];
  float cn = sigm_f(gF) * cp + sigm_f(gi) * tanhf(gg);
  float hn = sigm_f(go) * tanhf(cn);
  cs[bb * 128 + j] = cn;
  hs[bb * 128 + j] = (_Float16)hn;
  int tt = dir ? (TCONV - 1 - t) : t;
  size_t oidx = ((size_t)bb * TCONV + tt) * EMB + dir * NH + j;
  if (out32) out32[oidx] = hn;
  out16[oidx] = (_Float16)hn;
}

// ---------------------------------------------------------------------------
// MHA: scores = q @ k^T / 8 (K=64) per (b,head); qkv rows (b*128+t) x 768
// ---------------------------------------------------------------------------
__global__ void attn_scores_kernel(const _Float16* __restrict__ qkv, float* __restrict__ scores) {
  int bh = blockIdx.y;
  int b = bh >> 2, hh = bh & 3;
  int tile = blockIdx.x;
  int tm = tile >> 3, tn = tile & 7;
  int lane = threadIdx.x & 31;
  int lo = lane & 15, hi = lane >> 4;
  const _Float16* qb = qkv + (size_t)b * TCONV * 768 + hh * HD;
  const _Float16* kb = qb + 256;
  const _Float16* pA = qb + (size_t)(tm * 16 + lo) * 768 + hi * 8;
  const _Float16* pB = kb + (size_t)(tn * 16 + lo) * 768 + hi * 8;
  v8f c = {};
#pragma unroll
  for (int k0 = 0; k0 < HD; k0 += 32) {
    c = wmma_mac(wmma_load_a(pA + k0), wmma_load_b(pB + k0), c);
  }
  float* sb = scores + (size_t)bh * TCONV * TCONV;
#pragma unroll
  for (int v = 0; v < 8; ++v)
    sb[(size_t)(tm * 16 + hi * 8 + v) * TCONV + tn * 16 + lo] = c[v] * 0.125f;
}

__global__ void softmax_rows_kernel(const float* __restrict__ scores,
                                    _Float16* __restrict__ probs) {
  int row  = blockIdx.x;
  int lane = threadIdx.x & 31;
  const float* r = scores + (size_t)row * TCONV;
  float v[4];
  float m = -1e30f;
#pragma unroll
  for (int k = 0; k < 4; ++k) { v[k] = r[lane + (k << 5)]; m = fmaxf(m, v[k]); }
#pragma unroll
  for (int mm = 16; mm >= 1; mm >>= 1) m = fmaxf(m, __shfl_xor(m, mm, 32));
  float s = 0.f;
#pragma unroll
  for (int k = 0; k < 4; ++k) { v[k] = __expf(v[k] - m); s += v[k]; }
#pragma unroll
  for (int mm = 16; mm >= 1; mm >>= 1) s += __shfl_xor(s, mm, 32);
  float inv = 1.f / s;
#pragma unroll
  for (int k = 0; k < 4; ++k)
    probs[(size_t)row * TCONV + lane + (k << 5)] = (_Float16)(v[k] * inv);
}

// vT[(bh*64+d)*128 + t] = v from qkv — so probs@v uses K-contiguous B loads
__global__ void v_transpose_kernel(const _Float16* __restrict__ qkv, _Float16* __restrict__ vT) {
  int tid = blockIdx.x * blockDim.x + threadIdx.x;  // NT*256
  int d = tid & 63;
  int hh = (tid >> 6) & 3;
  int t = (tid >> 8) & 127;
  int b = tid >> 15;
  vT[(((size_t)b * 4 + hh) * 64 + d) * 128 + t] =
      qkv[((size_t)b * 128 + t) * 768 + 512 + hh * 64 + d];
}

// attn_out = probs @ v (K=128) -> f16 (b*128+t, 256)
__global__ void attn_av_kernel(const _Float16* __restrict__ probs,
                               const _Float16* __restrict__ vT,
                               _Float16* __restrict__ atto16) {
  int bh = blockIdx.y;
  int b = bh >> 2, hh = bh & 3;
  int tile = blockIdx.x;
  int tm = tile >> 2, tn = tile & 3;
  int lane = threadIdx.x & 31;
  int lo = lane & 15, hi = lane >> 4;
  const _Float16* pA = probs + (size_t)bh * TCONV * TCONV + (size_t)(tm * 16 + lo) * TCONV + hi * 8;
  const _Float16* pB = vT + (size_t)bh * HD * TCONV + (size_t)(tn * 16 + lo) * TCONV + hi * 8;
  v8f c = {};
#pragma unroll
  for (int k0 = 0; k0 < TCONV; k0 += 32) {
    c = wmma_mac(wmma_load_a(pA + k0), wmma_load_b(pB + k0), c);
  }
#pragma unroll
  for (int v = 0; v < 8; ++v) {
    int r  = tm * 16 + hi * 8 + v;
    int cc = tn * 16 + lo;
    atto16[((size_t)b * TCONV + r) * EMB + hh * HD + cc] = (_Float16)c[v];
  }
}

// ---------------------------------------------------------------------------
// pooling, FC head, utilities
// ---------------------------------------------------------------------------
__global__ void pool_kernel(const float* __restrict__ att, float* __restrict__ pooled) {
  int tid = blockIdx.x * blockDim.x + threadIdx.x;
  int b = tid >> 8, e = tid & 255;
  const float* p = att + (size_t)b * TCONV * EMB + e;
  float s = 0.f, mx = -1e30f;
  for (int t = 0; t < TCONV; ++t) {
    float v = p[(size_t)t * EMB];
    s += v;
    mx = fmaxf(mx, v);
  }
  pooled[b * 512 + e]       = s * (1.f / TCONV);
  pooled[b * 512 + 256 + e] = mx;
}

__global__ void fc_kernel(const float* __restrict__ pooled,
                          const float* __restrict__ w1, const float* __restrict__ b1,
                          const float* __restrict__ w2, const float* __restrict__ b2,
                          float* __restrict__ out) {
  __shared__ float z[128];
  int b = blockIdx.x, tid = threadIdx.x;
  const float* pr = pooled + (size_t)b * 512;
  float acc = b1[tid];
  for (int k = 0; k < 512; ++k) acc += pr[k] * w1[tid * 512 + k];
  z[tid] = fmaxf(acc, 0.f) * w2[tid];
  __syncthreads();
  for (int s = 64; s >= 1; s >>= 1) {
    if (tid < s) z[tid] += z[tid + s];
    __syncthreads();
  }
  if (tid == 0) out[b] = z[0] + b2[0];
}

__global__ void f32_to_f16_kernel(const float* __restrict__ s, _Float16* __restrict__ d, int n) {
  int i = blockIdx.x * blockDim.x + threadIdx.x;
  if (i < n) d[i] = (_Float16)s[i];
}

// transpose-convert: src (K,N) f32 -> dst (N,K) f16
__global__ void cvt_t_kernel(const float* __restrict__ s, _Float16* __restrict__ d, int K, int N) {
  int i = blockIdx.x * blockDim.x + threadIdx.x;
  if (i < K * N) {
    int k = i / N, n = i - k * N;
    d[(size_t)n * K + k] = (_Float16)s[i];
  }
}

__global__ void zero_kernel(unsigned int* __restrict__ p, int n) {
  int i = blockIdx.x * blockDim.x + threadIdx.x;
  if (i < n) p[i] = 0u;
}

// ---------------------------------------------------------------------------
// Host orchestration
// ---------------------------------------------------------------------------
extern "C" void kernel_launch(void* const* d_in, const int* in_sizes, int n_in,
                              void* d_out, int out_size, void* d_ws, size_t ws_size,
                              hipStream_t stream) {
  (void)in_sizes; (void)n_in; (void)out_size; (void)ws_size;
  const float* x        = (const float*)d_in[0];
  const float* conv1_w  = (const float*)d_in[1];
  const float* conv1_b  = (const float*)d_in[2];
  const float* bn1_g    = (const float*)d_in[3];
  const float* bn1_b    = (const float*)d_in[4];
  const float* conv2_w  = (const float*)d_in[5];
  const float* conv2_b  = (const float*)d_in[6];
  const float* bn2_g    = (const float*)d_in[7];
  const float* bn2_b    = (const float*)d_in[8];
  const float* g1_W     = (const float*)d_in[9];
  const float* g1_asrc  = (const float*)d_in[10];
  const float* g1_adst  = (const float*)d_in[11];
  const float* g1_adj   = (const float*)d_in[12];
  const float* n1_g     = (const float*)d_in[13];
  const float* n1_b     = (const float*)d_in[14];
  const float* g2_W     = (const float*)d_in[15];
  const float* g2_asrc  = (const float*)d_in[16];
  const float* g2_adst  = (const float*)d_in[17];
  const float* g2_adj   = (const float*)d_in[18];
  const float* n2_g     = (const float*)d_in[19];
  const float* n2_b     = (const float*)d_in[20];
  const float* l0f_Wih  = (const float*)d_in[21];
  const float* l0f_Whh  = (const float*)d_in[22];
  const float* l0f_bih  = (const float*)d_in[23];
  const float* l0f_bhh  = (const float*)d_in[24];
  const float* l0r_Wih  = (const float*)d_in[25];
  const float* l0r_Whh  = (const float*)d_in[26];
  const float* l0r_bih  = (const float*)d_in[27];
  const float* l0r_bhh  = (const float*)d_in[28];
  const float* l1f_Wih  = (const float*)d_in[29];
  const float* l1f_Whh  = (const float*)d_in[30];
  const float* l1f_bih  = (const float*)d_in[31];
  const float* l1f_bhh  = (const float*)d_in[32];
  const float* l1r_Wih  = (const float*)d_in[33];
  const float* l1r_Whh  = (const float*)d_in[34];
  const float* l1r_bih  = (const float*)d_in[35];
  const float* l1r_bhh  = (const float*)d_in[36];
  const float* mha_wqkv = (const float*)d_in[37];
  const float* mha_bqkv = (const float*)d_in[38];
  const float* mha_wo   = (const float*)d_in[39];
  const float* mha_bo   = (const float*)d_in[40];
  const float* an_g     = (const float*)d_in[41];
  const float* an_b     = (const float*)d_in[42];
  const float* pn_g     = (const float*)d_in[43];
  const float* pn_b     = (const float*)d_in[44];
  const float* fc1_w    = (const float*)d_in[45];
  const float* fc1_b    = (const float*)d_in[46];
  const float* fc2_w    = (const float*)d_in[47];
  const float* fc2_b    = (const float*)d_in[48];
  float* out = (float*)d_out;

  // ---- workspace carve (region reuse keeps total ~1.1 GB) ----
  char* ws = (char*)d_ws;
  size_t cur = 0;
  auto alloc = [&](size_t bytes) -> char* {
    char* p = ws + cur;
    cur += (bytes + 255) & ~(size_t)255;
    return p;
  };
  _Float16* g16    = (_Float16*)alloc((size_t)NCROWS * 64 * 2);   // conv out (reused: scores)
  float*    h1     = (float*)   alloc((size_t)NCROWS * 128 * 4);  // GAT GEMM out
  float*    x1     = (float*)   alloc((size_t)NCROWS * 128 * 4);  // GAT attn out
  _Float16* a16    = (_Float16*)alloc((size_t)NCROWS * 128 * 2);  // post-LN f16 (reused: seq16)
  float*    xp0f   = (float*)   alloc((size_t)NT * 512 * 4);      // reused: xproj layer1
  float*    xp0r   = (float*)   alloc((size_t)NT * 512 * 4);
  float*    gatesf = (float*)   alloc((size_t)256 * 512 * 4);
  float*    gatesr = (float*)   alloc((size_t)256 * 512 * 4);
  float*    cstf   = (float*)   alloc((size_t)256 * 128 * 4);     // states contiguous
  float*    cstr   = (float*)   alloc((size_t)256 * 128 * 4);
  _Float16* hstf   = (_Float16*)alloc((size_t)256 * 128 * 2);
  _Float16* hstr   = (_Float16*)alloc((size_t)256 * 128 * 2);
  _Float16* o0_16  = (_Float16*)alloc((size_t)NT * EMB * 2);
  _Float16* o1_16  = (_Float16*)alloc((size_t)NT * EMB * 2);
  float*    o1_32  = (float*)   alloc((size_t)NT * EMB * 4);
  _Float16* qkv16  = (_Float16*)alloc((size_t)NT * 768 * 2);
  _Float16* probs16= (_Float16*)alloc((size_t)1024 * TCONV * TCONV * 2);
  _Float16* vT     = (_Float16*)alloc((size_t)1024 * HD * TCONV * 2);
  _Float16* atto16 = (_Float16*)alloc((size_t)NT * EMB * 2);
  float*    ao32   = (float*)   alloc((size_t)NT * EMB * 4);
  float*    att32  = (float*)   alloc((size_t)NT * EMB * 4);
  float*    pooled = (float*)   alloc((size_t)256 * 512 * 4);
  float*    pooledln=(float*)   alloc((size_t)256 * 512 * 4);
  // f16 weights (all stored N x K)
  _Float16* w1t     = (_Float16*)alloc(128 * 64 * 2);
  _Float16* w2t     = (_Float16*)alloc(128 * 128 * 2);
  _Float16* wih0f16 = (_Float16*)alloc((size_t)512 * 2048 * 2);
  _Float16* whh0f16 = (_Float16*)alloc(512 * 128 * 2);
  _Float16* wih0r16 = (_Float16*)alloc((size_t)512 * 2048 * 2);
  _Float16* whh0r16 = (_Float16*)alloc(512 * 128 * 2);
  _Float16* wih1f16 = (_Float16*)alloc(512 * 256 * 2);
  _Float16* whh1f16 = (_Float16*)alloc(512 * 128 * 2);
  _Float16* wih1r16 = (_Float16*)alloc(512 * 256 * 2);
  _Float16* whh1r16 = (_Float16*)alloc(512 * 128 * 2);
  _Float16* wqkv16  = (_Float16*)alloc(768 * 256 * 2);
  _Float16* wo16    = (_Float16*)alloc(256 * 256 * 2);
  // aliases (lifetimes disjoint)
  float*    scores = (float*)g16;
  _Float16* seq16  = a16;
  float*    xp1f   = xp0f;
  float*    xp1r   = xp0r;

  // direct GEMM dispatch (template specialization -> branch-free K loop)
  auto gemmD = [&](const _Float16* A, int lda, const _Float16* Bm, int ldb,
                   const float* Cin, int ldc, const float* bias,
                   float* Dp, _Float16* D16, int ldd, int M, int N, int K) {
    int ntn = N >> 4;
    dim3 g((unsigned)((M >> 4) * ntn)), blk(32);
    if (Cin)
      gemm_wmma_kernel<1,0,1,0><<<g, blk, 0, stream>>>(A, lda, Bm, ldb, Cin, ldc, nullptr, Dp, nullptr, ldd, ntn, K);
    else if (bias && D16 && !Dp)
      gemm_wmma_kernel<0,1,0,1><<<g, blk, 0, stream>>>(A, lda, Bm, ldb, nullptr, 0, bias, nullptr, D16, ldd, ntn, K);
    else if (bias)
      gemm_wmma_kernel<0,1,1,0><<<g, blk, 0, stream>>>(A, lda, Bm, ldb, nullptr, 0, bias, Dp, nullptr, ldd, ntn, K);
    else
      gemm_wmma_kernel<0,0,1,0><<<g, blk, 0, stream>>>(A, lda, Bm, ldb, nullptr, 0, nullptr, Dp, nullptr, ldd, ntn, K);
  };
  // async LDS-staged GEMM (requires lda==ldb==K, M%32==0, N%64==0, K%64==0)
  auto gemmA = [&](const _Float16* A, const _Float16* Bm, float* Dp, int ldd,
                   int M, int N, int K) {
    int nbn = N / XBN;
    gemm_wmma_async_kernel<<<dim3((unsigned)((M / XBM) * nbn)), dim3(256), 0, stream>>>(
        A, Bm, Dp, ldd, nbn, K);
  };

  // ---- weight conversions (plain: already N x K; _t: transpose K x N -> N x K) ----
  cvt_t_kernel<<<dim3((64 * 128 + 255) / 256), dim3(256), 0, stream>>>(g1_W, w1t, 64, 128);
  cvt_t_kernel<<<dim3((128 * 128 + 255) / 256), dim3(256), 0, stream>>>(g2_W, w2t, 128, 128);
  struct Cvt { const float* s; _Float16* d; int n; };
  const Cvt cv[] = {
      {l0f_Wih, wih0f16, 512 * 2048}, {l0f_Whh, whh0f16, 512 * 128},
      {l0r_Wih, wih0r16, 512 * 2048}, {l0r_Whh, whh0r16, 512 * 128},
      {l1f_Wih, wih1f16, 512 * 256},  {l1f_Whh, whh1f16, 512 * 128},
      {l1r_Wih, wih1r16, 512 * 256},  {l1r_Whh, whh1r16, 512 * 128},
      {mha_wqkv, wqkv16, 768 * 256},  {mha_wo, wo16, 256 * 256},
  };
  for (int i = 0; i < (int)(sizeof(cv) / sizeof(cv[0])); ++i)
    f32_to_f16_kernel<<<dim3((cv[i].n + 255) / 256), dim3(256), 0, stream>>>(cv[i].s, cv[i].d, cv[i].n);

  // ---- conv stack ----
  conv_stage_kernel<<<dim3(NB * NCH * TCONV), dim3(64), 0, stream>>>(
      x, conv1_w, conv1_b, bn1_g, bn1_b, conv2_w, conv2_b, bn2_g, bn2_b, g16);

  // ---- GAT layer 1 (async LDS GEMM) ----
  gemmA(g16, w1t, h1, 128, NCROWS, 128, 64);
  gat_attn_kernel<<<dim3(NT), dim3(64), 0, stream>>>(h1, g1_asrc, g1_adst, g1_adj, x1);
  row_ln_kernel<<<dim3(NCROWS), dim3(32), 0, stream>>>(x1, nullptr, n1_g, n1_b, nullptr, a16, 128, 1);

  // ---- GAT layer 2 ----
  gemmA(a16, w2t, h1, 128, NCROWS, 128, 128);
  gat_attn_kernel<<<dim3(NT), dim3(64), 0, stream>>>(h1, g2_asrc, g2_adst, g2_adj, x1);
  row_ln_kernel<<<dim3(NCROWS), dim3(32), 0, stream>>>(x1, nullptr, n2_g, n2_b, nullptr, seq16, 128, 1);

  // ---- LSTM layer 0: x @ Wih^T via async GEMM (K=2048, the flagship) ----
  gemmA(seq16, wih0f16, xp0f, 512, NT, 512, 2048);
  gemmA(seq16, wih0r16, xp0r, 512, NT, 512, 2048);

  const int stateWords = (2 * 256 * 128 * 4 + 2 * 256 * 128 * 2) / 4;
  zero_kernel<<<dim3((stateWords + 255) / 256), dim3(256), 0, stream>>>((unsigned int*)cstf, stateWords);
  for (int t = 0; t < TCONV; ++t) {
    gemmD(hstf, 128, whh0f16, 128, xp0f + (size_t)t * 512, TCONV * 512, nullptr,
          gatesf, nullptr, 512, 256, 512, 128);
    gemmD(hstr, 128, whh0r16, 128, xp0r + (size_t)(TCONV - 1 - t) * 512, TCONV * 512, nullptr,
          gatesr, nullptr, 512, 256, 512, 128);
    lstm_cell_kernel<<<dim3(256), dim3(256), 0, stream>>>(
        gatesf, gatesr, l0f_bih, l0f_bhh, l0r_bih, l0r_bhh,
        cstf, cstr, hstf, hstr, nullptr, o0_16, t);
  }

  // ---- LSTM layer 1 ----
  gemmA(o0_16, wih1f16, xp1f, 512, NT, 512, 256);
  gemmA(o0_16, wih1r16, xp1r, 512, NT, 512, 256);
  zero_kernel<<<dim3((stateWords + 255) / 256), dim3(256), 0, stream>>>((unsigned int*)cstf, stateWords);
  for (int t = 0; t < TCONV; ++t) {
    gemmD(hstf, 128, whh1f16, 128, xp1f + (size_t)t * 512, TCONV * 512, nullptr,
          gatesf, nullptr, 512, 256, 512, 128);
    gemmD(hstr, 128, whh1r16, 128, xp1r + (size_t)(TCONV - 1 - t) * 512, TCONV * 512, nullptr,
          gatesr, nullptr, 512, 256, 512, 128);
    lstm_cell_kernel<<<dim3(256), dim3(256), 0, stream>>>(
        gatesf, gatesr, l1f_bih, l1f_bhh, l1r_bih, l1r_bhh,
        cstf, cstr, hstf, hstr, o1_32, o1_16, t);
  }

  // ---- MHA ----
  gemmD(o1_16, 256, wqkv16, 256, nullptr, 0, mha_bqkv, nullptr, qkv16, 768, NT, 768, 256);
  attn_scores_kernel<<<dim3(64, NB * NHEADS), dim3(32), 0, stream>>>(qkv16, scores);
  softmax_rows_kernel<<<dim3(NB * NHEADS * TCONV), dim3(32), 0, stream>>>(scores, probs16);
  v_transpose_kernel<<<dim3(NT), dim3(256), 0, stream>>>(qkv16, vT);
  attn_av_kernel<<<dim3(32, NB * NHEADS), dim3(32), 0, stream>>>(probs16, vT, atto16);
  gemmD(atto16, 256, wo16, 256, nullptr, 0, mha_bo, ao32, nullptr, 256, NT, 256, 256);
  row_ln_kernel<<<dim3(NT), dim3(32), 0, stream>>>(ao32, o1_32, an_g, an_b, att32, nullptr, EMB, 0);

  // ---- pooling + head ----
  pool_kernel<<<dim3(256), dim3(256), 0, stream>>>(att32, pooled);
  row_ln_kernel<<<dim3(NB), dim3(32), 0, stream>>>(pooled, nullptr, pn_g, pn_b, pooledln, nullptr, 512, 0);
  fc_kernel<<<dim3(NB), dim3(128), 0, stream>>>(pooledln, fc1_w, fc1_b, fc2_w, fc2_b, out);
}
